// LlamaMLP_7267084665263
// MI455X (gfx1250) — matrix operand order repaired and checked by reading, not compiled, not run on hardware
//
#include <hip/hip_runtime.h>
#include <hip/hip_bf16.h>

// ---------------------------------------------------------------------------
// DeepSeek-style MoE MLP for MI455X (gfx1250, wave32, WMMA + async-to-LDS).
// Dense evaluation of all experts (combine weight is 0 for unselected experts,
// so this is exact); f16 WMMA GEMMs, fp32 accum, LDS double-buffered tiles
// filled with GLOBAL_LOAD_ASYNC_TO_LDS_B128 (ASYNCcnt-tracked).
// Inner loop: issue all ds_load_b128 fragments, wait once, then a dense
// back-to-back WMMA block (keeps the XDL pipe saturated).
// ---------------------------------------------------------------------------

typedef __attribute__((ext_vector_type(16))) _Float16 v16h;
typedef __attribute__((ext_vector_type(8)))  _Float16 v8h;
typedef __attribute__((ext_vector_type(8)))  float    v8f;

constexpr int DIM  = 1024;
constexpr int HID  = 2816;
constexpr int NEXP = 7;
constexpr int TOKS = 4 * 2048;   // B*S

constexpr int KSTEP = 32;        // K per stage
constexpr int ROW_B = 80;        // padded LDS row stride (bytes): 20-bank stride,
                                 // gcd(5,16)=1 -> conflict-free b128 reads
constexpr unsigned BUF_B = 128 * ROW_B;   // bytes per LDS tile buffer

// ---------------------------------------------------------------------------
// CDNA5 async copy global -> LDS (16B per lane), tracked with ASYNCcnt
// ---------------------------------------------------------------------------
__device__ __forceinline__ void async_copy_b128(unsigned lds_addr, const void* gaddr) {
  asm volatile("global_load_async_to_lds_b128 %0, %1, off"
               :: "v"(lds_addr), "v"(gaddr)
               : "memory");
}

__device__ __forceinline__ void wait_async0() {
#if __has_builtin(__builtin_amdgcn_s_wait_asynccnt)
  __builtin_amdgcn_s_wait_asynccnt(0);
#else
  asm volatile("s_wait_asynccnt 0x0" ::: "memory");
#endif
}

// ---------------------------------------------------------------------------
// LDS fragment loads (wave32 WMMA operand layouts)
// A (16x32, M per lane): elems 0-7 at K=8g, elems 8-15 at K=16+8g
// B (32x16, N per lane): elems 0-7 at K=8g, elems 8-15 at K=16+8g
// ---------------------------------------------------------------------------
__device__ __forceinline__ v16h lds_fragA(const _Float16* base, int row, int lg) {
  const char* p = (const char*)base + row * ROW_B + 16 * lg;
  v8h lo = *(const v8h*)(p);
  v8h hi = *(const v8h*)(p + 32);
  return __builtin_shufflevector(lo, hi, 0,1,2,3,4,5,6,7,8,9,10,11,12,13,14,15);
}

__device__ __forceinline__ v16h lds_fragB(const _Float16* base, int row, int lg) {
  const char* p = (const char*)base + row * ROW_B + 16 * lg;
  v8h lo = *(const v8h*)(p);
  v8h hi = *(const v8h*)(p + 32);
  return __builtin_shufflevector(lo, hi, 0,1,2,3,4,5,6,7,8,9,10,11,12,13,14,15);
}

__device__ __forceinline__ v8f wmma16(v16h a, v16h b, v8f c) {
  return __builtin_amdgcn_wmma_f32_16x16x32_f16(
      false, a, false, b, (short)0, c, false, false);
}

// ---------------------------------------------------------------------------
// elementwise fp32 -> f16 convert (for x)
// ---------------------------------------------------------------------------
__global__ void convert_f32_to_f16(const float* __restrict__ in,
                                   _Float16* __restrict__ out, int n) {
  int i = blockIdx.x * blockDim.x + threadIdx.x;
  if (i < n) out[i] = (_Float16)in[i];
}

// ---------------------------------------------------------------------------
// tiled transpose + convert:  in fp32 [R][C]  ->  out f16 [C][R]
// ---------------------------------------------------------------------------
__global__ void transpose_f32_to_f16(const float* __restrict__ in,
                                     _Float16* __restrict__ out,
                                     int R, int C) {
  __shared__ float tile[32][33];
  const int tx = threadIdx.x, ty = threadIdx.y;
  const int c0 = blockIdx.x * 32, r0 = blockIdx.y * 32;
#pragma unroll
  for (int j = 0; j < 32; j += 8)
    tile[ty + j][tx] = in[(size_t)(r0 + ty + j) * C + (c0 + tx)];
  __syncthreads();
#pragma unroll
  for (int j = 0; j < 32; j += 8)
    out[(size_t)(c0 + ty + j) * R + (r0 + tx)] = (_Float16)tile[tx][ty + j];
}

// ---------------------------------------------------------------------------
// router: logits = (x @ Wr) * bias ; probs = sigmoid ; top-2 ; normalize
// ---------------------------------------------------------------------------
__global__ void router_kernel(const float* __restrict__ x,
                              const float* __restrict__ Wr,   // [DIM][NEXP]
                              const float* __restrict__ bias, // [NEXP]
                              float* __restrict__ rw) {       // [TOKS][NEXP]
  const int lane = threadIdx.x & 31;
  const int wave = threadIdx.x >> 5;
  const int t = blockIdx.x * 8 + wave;
  if (t >= TOKS) return;

  float acc[NEXP];
#pragma unroll
  for (int e = 0; e < NEXP; ++e) acc[e] = 0.0f;

  const float* xr = x + (size_t)t * DIM;
  for (int d = lane; d < DIM; d += 32) {
    const float xv = xr[d];
    const float* w = Wr + (size_t)d * NEXP;
#pragma unroll
    for (int e = 0; e < NEXP; ++e) acc[e] += xv * w[e];
  }
#pragma unroll
  for (int e = 0; e < NEXP; ++e) {
#pragma unroll
    for (int off = 16; off >= 1; off >>= 1)
      acc[e] += __shfl_down(acc[e], off, 32);
  }
  if (lane == 0) {
    float p[NEXP];
#pragma unroll
    for (int e = 0; e < NEXP; ++e) {
      const float lg = acc[e] * bias[e];
      p[e] = 1.0f / (1.0f + __expf(-lg));
    }
    int i1 = 0;
#pragma unroll
    for (int e = 1; e < NEXP; ++e) if (p[e] > p[i1]) i1 = e;
    int i2 = (i1 == 0) ? 1 : 0;
#pragma unroll
    for (int e = 0; e < NEXP; ++e)
      if (e != i1 && e != i2 && p[e] > p[i2]) i2 = e;
    const float s = p[i1] + p[i2];
    float o[NEXP];
#pragma unroll
    for (int e = 0; e < NEXP; ++e) o[e] = 0.0f;
    o[i1] = p[i1] / s;
    o[i2] = p[i2] / s;
#pragma unroll
    for (int e = 0; e < NEXP; ++e) rw[(size_t)t * NEXP + e] = o[e];
  }
}

// ---------------------------------------------------------------------------
// fused gate/up GEMM + silu*mul epilogue, LDS double-buffered, async staged.
//   X    : [TOKS][DIM] f16 ; Wg,Wu : [HID][DIM] f16 (transposed) ;
//   H    : [TOKS][HID] f16 = silu(X@Wg) * (X@Wu)
// block = 8 waves (4x2), wave tile 32x64, block tile 128x128, K-step 32
// ---------------------------------------------------------------------------
__global__ __launch_bounds__(256)
void gateup_kernel(const _Float16* __restrict__ X,
                   const _Float16* __restrict__ Wg,
                   const _Float16* __restrict__ Wu,
                   _Float16* __restrict__ H) {
  __shared__ __align__(16) _Float16 sX[2][BUF_B / 2];
  __shared__ __align__(16) _Float16 sG[2][BUF_B / 2];
  __shared__ __align__(16) _Float16 sU[2][BUF_B / 2];

  const int tid  = threadIdx.x;
  const int lane = tid & 31;
  const int wave = tid >> 5;
  const int wm = wave & 3;             // 0..3
  const int wn = wave >> 2;            // 0..1
  const int m_blk = blockIdx.y * 128;
  const int n_blk = blockIdx.x * 128;
  const int lr = lane & 15;
  const int lg = lane >> 4;

  // ---- per-thread staging state: 2 chunks of 16B per tile per K-step ----
  const int r0  = tid >> 2;            // 0..63 ; chunk i covers row r0 + 64*i
  const int off = (tid & 3) * 16;      // byte offset within the 64B row
  const char* gx[2]; const char* gg[2]; const char* gu[2];
  unsigned lx[2], lG[2], lU[2];
#pragma unroll
  for (int i = 0; i < 2; ++i) {
    const int row = r0 + 64 * i;
    gx[i] = (const char*)(X  + (size_t)(m_blk + row) * DIM) + off;
    gg[i] = (const char*)(Wg + (size_t)(n_blk + row) * DIM) + off;
    gu[i] = (const char*)(Wu + (size_t)(n_blk + row) * DIM) + off;
    const unsigned l = (unsigned)(row * ROW_B + off);
    lx[i] = (unsigned)(size_t)(&sX[0][0]) + l;
    lG[i] = (unsigned)(size_t)(&sG[0][0]) + l;
    lU[i] = (unsigned)(size_t)(&sU[0][0]) + l;
  }
  auto stage = [&](int buf) {
    const unsigned b = (unsigned)buf * BUF_B;
#pragma unroll
    for (int i = 0; i < 2; ++i) {
      async_copy_b128(lx[i] + b, gx[i]); gx[i] += KSTEP * 2;
      async_copy_b128(lG[i] + b, gg[i]); gg[i] += KSTEP * 2;
      async_copy_b128(lU[i] + b, gu[i]); gu[i] += KSTEP * 2;
    }
  };

  v8f accg[2][4] = {};
  v8f accu[2][4] = {};

  stage(0);
  wait_async0();
  __syncthreads();

  int cur = 0;
  for (int k0 = 0; k0 < DIM; k0 += KSTEP) {
    if (k0 + KSTEP < DIM) stage(cur ^ 1);   // overlap next copy with compute

    // hoist ALL fragments, then a dense WMMA block
    v16h a[2], bg[4], bu[4];
#pragma unroll
    for (int mi = 0; mi < 2; ++mi)
      a[mi] = lds_fragA(sX[cur], wm * 32 + mi * 16 + lr, lg);
#pragma unroll
    for (int ni = 0; ni < 4; ++ni)
      bg[ni] = lds_fragB(sG[cur], wn * 64 + ni * 16 + lr, lg);
#pragma unroll
    for (int ni = 0; ni < 4; ++ni)
      bu[ni] = lds_fragB(sU[cur], wn * 64 + ni * 16 + lr, lg);

#pragma unroll
    for (int ni = 0; ni < 4; ++ni)
#pragma unroll
      for (int mi = 0; mi < 2; ++mi) {
        accg[mi][ni] = wmma16(a[mi], bg[ni], accg[mi][ni]);
        accu[mi][ni] = wmma16(a[mi], bu[ni], accu[mi][ni]);
      }

    wait_async0();                     // next buffer resident
    __syncthreads();                   // all waves done reading cur
    cur ^= 1;
  }

  // epilogue: h = silu(g) * u
#pragma unroll
  for (int mi = 0; mi < 2; ++mi) {
#pragma unroll
    for (int ni = 0; ni < 4; ++ni) {
#pragma unroll
      for (int r = 0; r < 8; ++r) {
        const float g = accg[mi][ni][r];
        const float u = accu[mi][ni][r];
        const float h = (g / (1.0f + __expf(-g))) * u;
        const int row = m_blk + wm * 32 + mi * 16 + r + 8 * lg;
        const int col = n_blk + wn * 64 + ni * 16 + lr;
        H[(size_t)row * HID + col] = (_Float16)h;
      }
    }
  }
}

// ---------------------------------------------------------------------------
// down-projection GEMM with per-token scale accumulation into fp32 out.
//   Hm : [TOKS][HID] f16 ; Wd : [DIM][HID] f16 (transposed) ;
//   out: [TOKS][DIM] f32 ; INIT: out = Y, else out += Y * rw[t][expert]
// ---------------------------------------------------------------------------
template <bool INIT>
__global__ __launch_bounds__(256)
void down_kernel(const _Float16* __restrict__ Hm,
                 const _Float16* __restrict__ Wd,
                 const float* __restrict__ rw, int expert,
                 float* __restrict__ out) {
  __shared__ __align__(16) _Float16 sA[2][BUF_B / 2];
  __shared__ __align__(16) _Float16 sB[2][BUF_B / 2];

  const int tid  = threadIdx.x;
  const int lane = tid & 31;
  const int wave = tid >> 5;
  const int wm = wave & 3;
  const int wn = wave >> 2;
  const int m_blk = blockIdx.y * 128;
  const int n_blk = blockIdx.x * 128;
  const int lr = lane & 15;
  const int lg = lane >> 4;

  const int r0  = tid >> 2;
  const int off = (tid & 3) * 16;
  const char* ga[2]; const char* gb[2];
  unsigned lA[2], lB[2];
#pragma unroll
  for (int i = 0; i < 2; ++i) {
    const int row = r0 + 64 * i;
    ga[i] = (const char*)(Hm + (size_t)(m_blk + row) * HID) + off;
    gb[i] = (const char*)(Wd + (size_t)(n_blk + row) * HID) + off;
    const unsigned l = (unsigned)(row * ROW_B + off);
    lA[i] = (unsigned)(size_t)(&sA[0][0]) + l;
    lB[i] = (unsigned)(size_t)(&sB[0][0]) + l;
  }
  auto stage = [&](int buf) {
    const unsigned b = (unsigned)buf * BUF_B;
#pragma unroll
    for (int i = 0; i < 2; ++i) {
      async_copy_b128(lA[i] + b, ga[i]); ga[i] += KSTEP * 2;
      async_copy_b128(lB[i] + b, gb[i]); gb[i] += KSTEP * 2;
    }
  };

  v8f acc[2][4] = {};

  stage(0);
  wait_async0();
  __syncthreads();

  int cur = 0;
  for (int k0 = 0; k0 < HID; k0 += KSTEP) {
    if (k0 + KSTEP < HID) stage(cur ^ 1);

    v16h a[2], b[4];
#pragma unroll
    for (int mi = 0; mi < 2; ++mi)
      a[mi] = lds_fragA(sA[cur], wm * 32 + mi * 16 + lr, lg);
#pragma unroll
    for (int ni = 0; ni < 4; ++ni)
      b[ni] = lds_fragB(sB[cur], wn * 64 + ni * 16 + lr, lg);

#pragma unroll
    for (int ni = 0; ni < 4; ++ni)
#pragma unroll
      for (int mi = 0; mi < 2; ++mi)
        acc[mi][ni] = wmma16(a[mi], b[ni], acc[mi][ni]);

    wait_async0();
    __syncthreads();
    cur ^= 1;
  }

  float sc[2][8];
  if (!INIT) {
#pragma unroll
    for (int mi = 0; mi < 2; ++mi)
#pragma unroll
      for (int r = 0; r < 8; ++r)
        sc[mi][r] = rw[(size_t)(m_blk + wm * 32 + mi * 16 + r + 8 * lg) * NEXP + expert];
  }

#pragma unroll
  for (int mi = 0; mi < 2; ++mi) {
#pragma unroll
    for (int ni = 0; ni < 4; ++ni) {
#pragma unroll
      for (int r = 0; r < 8; ++r) {
        const int row = m_blk + wm * 32 + mi * 16 + r + 8 * lg;
        const int col = n_blk + wn * 64 + ni * 16 + lr;
        const size_t idx = (size_t)row * DIM + col;
        if (INIT) out[idx] = acc[mi][ni][r];
        else      out[idx] += acc[mi][ni][r] * sc[mi][r];
      }
    }
  }
}

// ---------------------------------------------------------------------------
// launch
// ---------------------------------------------------------------------------
extern "C" void kernel_launch(void* const* d_in, const int* in_sizes, int n_in,
                              void* d_out, int out_size, void* d_ws, size_t ws_size,
                              hipStream_t stream) {
  (void)in_sizes; (void)n_in; (void)out_size; (void)ws_size;

  const float* x  = (const float*)d_in[0];
  const float* sg = (const float*)d_in[1];
  const float* su = (const float*)d_in[2];
  const float* sd = (const float*)d_in[3];
  const float* rg = (const float*)d_in[4];
  const float* ru = (const float*)d_in[5];
  const float* rd = (const float*)d_in[6];
  const float* Wr = (const float*)d_in[7];
  const float* rb = (const float*)d_in[8];
  float* out = (float*)d_out;

  char* ws = (char*)d_ws;
  constexpr size_t OFF_X  = 0;
  constexpr size_t SZ_X   = (size_t)TOKS * DIM * 2;
  constexpr size_t OFF_WG = OFF_X + SZ_X;
  constexpr size_t SZ_W   = (size_t)DIM * HID * 2;
  constexpr size_t OFF_WU = OFF_WG + SZ_W;
  constexpr size_t OFF_WD = OFF_WU + SZ_W;
  constexpr size_t OFF_H  = OFF_WD + SZ_W;
  constexpr size_t SZ_H   = (size_t)TOKS * HID * 2;
  constexpr size_t OFF_RW = OFF_H + SZ_H;

  _Float16* Xh  = (_Float16*)(ws + OFF_X);
  _Float16* Wgh = (_Float16*)(ws + OFF_WG);   // [HID][DIM]
  _Float16* Wuh = (_Float16*)(ws + OFF_WU);   // [HID][DIM]
  _Float16* Wdh = (_Float16*)(ws + OFF_WD);   // [DIM][HID]
  _Float16* Hb  = (_Float16*)(ws + OFF_H);    // [TOKS][HID]
  float*    rwb = (float*)   (ws + OFF_RW);   // [TOKS][NEXP]

  const dim3 tb_tr(32, 8);
  const dim3 gr_gu(HID / 32, DIM / 32);
  const dim3 gr_dn(DIM / 32, HID / 32);
  const dim3 g_gateup(HID / 128, TOKS / 128);
  const dim3 g_down(DIM / 128, TOKS / 128);

  convert_f32_to_f16<<<(TOKS * DIM) / 256, 256, 0, stream>>>(x, Xh, TOKS * DIM);
  router_kernel<<<TOKS / 8, 256, 0, stream>>>(x, Wr, rb, rwb);

  // shared expert: initializes out
  transpose_f32_to_f16<<<gr_gu, tb_tr, 0, stream>>>(sg, Wgh, DIM, HID);
  transpose_f32_to_f16<<<gr_gu, tb_tr, 0, stream>>>(su, Wuh, DIM, HID);
  transpose_f32_to_f16<<<gr_dn, tb_tr, 0, stream>>>(sd, Wdh, HID, DIM);
  gateup_kernel<<<g_gateup, 256, 0, stream>>>(Xh, Wgh, Wuh, Hb);
  down_kernel<true><<<g_down, 256, 0, stream>>>(Hb, Wdh, nullptr, 0, out);

  // routed experts: dense, scaled by per-token combine weight (0 => no-op)
  for (int e = 0; e < NEXP; ++e) {
    transpose_f32_to_f16<<<gr_gu, tb_tr, 0, stream>>>(rg + (size_t)e * DIM * HID, Wgh, DIM, HID);
    transpose_f32_to_f16<<<gr_gu, tb_tr, 0, stream>>>(ru + (size_t)e * DIM * HID, Wuh, DIM, HID);
    transpose_f32_to_f16<<<gr_dn, tb_tr, 0, stream>>>(rd + (size_t)e * HID * DIM, Wdh, HID, DIM);
    gateup_kernel<<<g_gateup, 256, 0, stream>>>(Xh, Wgh, Wuh, Hb);
    down_kernel<false><<<g_down, 256, 0, stream>>>(Hb, Wdh, rwb, e, out);
  }
}
